// Transformer_30356828848528
// MI455X (gfx1250) — matrix operand order repaired and checked by reading, not compiled, not run on hardware
//
#include <hip/hip_runtime.h>
#include <hip/hip_bf16.h>
#include <math.h>

typedef _Float16 half_t;
typedef __attribute__((ext_vector_type(16))) _Float16 v16h;
typedef __attribute__((ext_vector_type(8)))  _Float16 v8h;
typedef __attribute__((ext_vector_type(8)))  float    v8f;

#define DIM   1024
#define SEQ   2048
#define BATCH 2
#define TOK   (BATCH * SEQ)   // 4096 tokens
#define QKV3  (3 * DIM)       // 3072
#define HEADS 16
#define DHEAD 64

// ---------------------------------------------------------------------------
// WMMA helpers (CDNA5 v_wmma_f32_16x16x32_f16, wave32)
// A order used here (16x32 f16): lane r=ln&15 holds row M=r; halves 0..15 -> K=16*g+0..15
// (g=ln>>4)  => one contiguous 32B chunk.
// B order used here (32x16 f16): lane holds column N=ln&15; halves 0..15 -> K=16*g+0..15
// => one contiguous 32B chunk when reading B^T ([N,K] row-major).
// C/D layout (16x16 f32): vgpr v, lanes 0-15 -> M=v, lanes 16-31 -> M=v+8; N=ln&15.
// ---------------------------------------------------------------------------
static __device__ __forceinline__ v16h load_afrag(const half_t* p) {
  return *(const v16h*)p;
}
static __device__ __forceinline__ v16h load_bfrag(const half_t* p) {
  return *(const v16h*)p;
}
static __device__ __forceinline__ v8f wmma_f16(v16h a, v16h b, v8f c) {
  return __builtin_amdgcn_wmma_f32_16x16x32_f16(false, a, false, b, (short)0, c,
                                                false, false);
}

// ---------------------------------------------------------------------------
// Weight convert + transpose: w [K,N] fp32 (row-major) -> wt [N,K] f16
// ---------------------------------------------------------------------------
__global__ __launch_bounds__(256) void convT_kernel(const float* __restrict__ w,
                                                    half_t* __restrict__ wt,
                                                    int K, int N) {
  size_t id = (size_t)blockIdx.x * 256 + threadIdx.x;
  if (id >= (size_t)K * N) return;
  int n = (int)(id % N);
  int k = (int)(id / N);
  wt[(size_t)n * K + k] = (half_t)w[id];
}

// ---------------------------------------------------------------------------
// V transpose: qkv [TOK, 3072] f16 (V at col 2048 + h*64 + d) ->
//              vT [b, h, d, key] f16 (key fastest => coalesced writes, and
//              contiguous 32B B-fragment loads in the attention P*V stage)
// ---------------------------------------------------------------------------
__global__ __launch_bounds__(256) void vtrans_kernel(const half_t* __restrict__ qkv,
                                                     half_t* __restrict__ vT) {
  size_t id = (size_t)blockIdx.x * 256 + threadIdx.x;
  if (id >= (size_t)TOK * DIM) return;
  int key    = (int)(id & (SEQ - 1));
  size_t t   = id >> 11;              // (b*HEADS + h)*DHEAD + d
  int d      = (int)(t & (DHEAD - 1));
  size_t bh  = t >> 6;                // b*HEADS + h
  int h      = (int)(bh & (HEADS - 1));
  int b      = (int)(bh >> 4);
  vT[id] = qkv[((size_t)b * SEQ + key) * QKV3 + 2 * DIM + h * DHEAD + d];
}

// ---------------------------------------------------------------------------
// LayerNorm over last dim (1024), fp32 in -> f16 out. One block per row.
// ---------------------------------------------------------------------------
__global__ __launch_bounds__(256) void ln_kernel(const float* __restrict__ x,
                                                 const float* __restrict__ g,
                                                 const float* __restrict__ b,
                                                 half_t* __restrict__ out) {
  int row = blockIdx.x;
  const float* xr = x + (size_t)row * DIM;
  float vals[4];
  float s = 0.f, ss = 0.f;
#pragma unroll
  for (int i = 0; i < 4; ++i) {
    float v = xr[threadIdx.x + i * 256];
    vals[i] = v; s += v; ss += v * v;
  }
#pragma unroll
  for (int m = 1; m < 32; m <<= 1) {
    s  += __shfl_xor(s,  m, 32);
    ss += __shfl_xor(ss, m, 32);
  }
  __shared__ float sh_s[8], sh_q[8];
  if ((threadIdx.x & 31) == 0) { sh_s[threadIdx.x >> 5] = s; sh_q[threadIdx.x >> 5] = ss; }
  __syncthreads();
  s = 0.f; ss = 0.f;
#pragma unroll
  for (int w = 0; w < 8; ++w) { s += sh_s[w]; ss += sh_q[w]; }
  float mu  = s * (1.0f / DIM);
  float var = ss * (1.0f / DIM) - mu * mu;
  float inv = rsqrtf(var + 1e-5f);
  half_t* orow = out + (size_t)row * DIM;
#pragma unroll
  for (int i = 0; i < 4; ++i) {
    int c = threadIdx.x + i * 256;
    orow[c] = (half_t)((vals[i] - mu) * inv * g[c] + b[c]);
  }
}

// ---------------------------------------------------------------------------
// WMMA GEMM: C[M,N] (f32 acc) = A[M,K] f16  x  Bt[N,K] f16 (B transposed)
// Block = 128 thr (4 waves). Each wave owns a 64x64 tile (4x4 WMMA grid,
// 16 v8f accumulators, 4x fragment reuse); block covers 128x128.
// Epilogue: +bias, optional exact GELU, +fp32 residual, store f32 and/or f16.
// ---------------------------------------------------------------------------
__global__ __launch_bounds__(128) void gemm_kernel(
    const half_t* __restrict__ A, const half_t* __restrict__ Bt,
    float* __restrict__ outF, half_t* __restrict__ outH,
    const float* __restrict__ bias, const float* __restrict__ res,
    int M, int N, int K, int gelu) {
  int wave = threadIdx.x >> 5, lane = threadIdx.x & 31;
  int r = lane & 15, g = lane >> 4;
  int row0 = blockIdx.y * 128 + (wave >> 1) * 64;
  int col0 = blockIdx.x * 128 + (wave & 1) * 64;

  v8f acc[4][4];
#pragma unroll
  for (int i = 0; i < 4; ++i)
#pragma unroll
    for (int j = 0; j < 4; ++j) acc[i][j] = v8f{};

  const half_t* Abase = A  + (size_t)(row0 + r) * K + 16 * g;
  const half_t* Bbase = Bt + (size_t)(col0 + r) * K + 16 * g;
  const size_t Arow16 = (size_t)16 * K;

  for (int k0 = 0; k0 < K; k0 += 32) {
    v16h af[4], bf[4];
#pragma unroll
    for (int i = 0; i < 4; ++i) af[i] = load_afrag(Abase + (size_t)i * Arow16 + k0);
#pragma unroll
    for (int j = 0; j < 4; ++j) bf[j] = load_bfrag(Bbase + (size_t)j * Arow16 + k0);
#pragma unroll
    for (int i = 0; i < 4; ++i)
#pragma unroll
      for (int j = 0; j < 4; ++j) acc[i][j] = wmma_f16(af[i], bf[j], acc[i][j]);
  }

#pragma unroll
  for (int i = 0; i < 4; ++i) {
#pragma unroll
    for (int j = 0; j < 4; ++j) {
#pragma unroll
      for (int v = 0; v < 8; ++v) {
        int m = row0 + 16 * i + v + 8 * g;
        int n = col0 + 16 * j + r;
        float val = acc[i][j][v];
        if (bias) val += bias[n];
        if (gelu) val = 0.5f * val * (1.0f + erff(val * 0.70710678118f));
        size_t idx = (size_t)m * N + n;
        if (res)  val += res[idx];
        if (outF) outF[idx] = val;
        if (outH) outH[idx] = (half_t)val;
      }
    }
  }
}

// ---------------------------------------------------------------------------
// Flash attention: one wave = one 16-row query tile of one (b,h).
// Q,K read from qkv f16 [TOK, 3072]; V read pre-transposed from vT [b,h,d,key].
// Online softmax over key blocks of 32; P converted C-layout -> A-fragment
// through a per-wave 1KB LDS region. 8 WMMA per 32-key block.
// ---------------------------------------------------------------------------
__global__ __launch_bounds__(128) void attn_kernel(const half_t* __restrict__ qkv,
                                                   const half_t* __restrict__ vT,
                                                   half_t* __restrict__ attn_out) {
  __shared__ half_t lds[4 * 512];   // 4 waves x (16 rows x 32 keys) f16
  int wave = threadIdx.x >> 5, lane = threadIdx.x & 31;
  int r = lane & 15, g = lane >> 4;
  int task = blockIdx.x * 4 + wave;       // 0..4095
  int qt = task & 127;                    // query tile (128 per head)
  int h  = (task >> 7) & 15;
  int b  = task >> 11;

  const size_t rs = QKV3;
  const half_t* Qb  = qkv + (size_t)b * SEQ * rs + h * DHEAD;
  const half_t* Kb  = Qb + DIM;
  const half_t* VTb = vT + (((size_t)b * HEADS + h) * DHEAD) * SEQ;
  int q0 = qt * 16;

  // Q fragments for d=0..31 and d=32..63, resident for the whole key loop
  v16h qa0 = load_afrag(Qb + (size_t)(q0 + r) * rs + 0  + 16 * g);
  v16h qa1 = load_afrag(Qb + (size_t)(q0 + r) * rs + 32 + 16 * g);

  float rm[8], rl[8];
#pragma unroll
  for (int v = 0; v < 8; ++v) { rm[v] = -1e30f; rl[v] = 0.f; }
  v8f o0 = {}, o1 = {}, o2 = {}, o3 = {};
  half_t* P = lds + wave * 512;
  const float scale = 0.125f;             // 64^-0.5

  for (int j0 = 0; j0 < SEQ; j0 += 32) {
    // S tiles: s0 = keys j0..j0+15, s1 = keys j0+16..j0+31 (K-dim = 64)
    v8f s0 = {}, s1 = {};
    {
      v16h kb;
      kb = load_bfrag(Kb + (size_t)(j0 + r)      * rs + 0  + 16 * g); s0 = wmma_f16(qa0, kb, s0);
      kb = load_bfrag(Kb + (size_t)(j0 + r)      * rs + 32 + 16 * g); s0 = wmma_f16(qa1, kb, s0);
      kb = load_bfrag(Kb + (size_t)(j0 + 16 + r) * rs + 0  + 16 * g); s1 = wmma_f16(qa0, kb, s1);
      kb = load_bfrag(Kb + (size_t)(j0 + 16 + r) * rs + 32 + 16 * g); s1 = wmma_f16(qa1, kb, s1);
    }

    float p0[8], p1[8], corr[8];
#pragma unroll
    for (int v = 0; v < 8; ++v) {
      float a = s0[v] * scale, c = s1[v] * scale;
      float bm = fmaxf(a, c);
#pragma unroll
      for (int m = 1; m < 16; m <<= 1) bm = fmaxf(bm, __shfl_xor(bm, m, 32));
      float mn = fmaxf(rm[v], bm);
      corr[v] = expf(rm[v] - mn);
      float e0 = expf(a - mn), e1 = expf(c - mn);
      p0[v] = e0; p1[v] = e1;
      float sum = e0 + e1;
#pragma unroll
      for (int m = 1; m < 16; m <<= 1) sum += __shfl_xor(sum, m, 32);
      rl[v] = rl[v] * corr[v] + sum;
      rm[v] = mn;
    }
#pragma unroll
    for (int v = 0; v < 8; ++v) {
      o0[v] *= corr[v]; o1[v] *= corr[v]; o2[v] *= corr[v]; o3[v] *= corr[v];
    }

    // P (C layout) -> LDS row-major [16 rows][32 keys]
#pragma unroll
    for (int v = 0; v < 8; ++v) {
      P[(v + 8 * g) * 32 + r]      = (half_t)p0[v];
      P[(v + 8 * g) * 32 + 16 + r] = (half_t)p1[v];
    }
    __syncthreads();
    v16h pa = load_afrag(P + r * 32 + 16 * g);   // A-fragment of P (16x32)

    // O += P x V ; V B-fragments are contiguous 32B reads from vT [d, key]
    {
      v16h vb;
      vb = load_bfrag(VTb + (size_t)(0 * 16 + r) * SEQ + j0 + 16 * g); o0 = wmma_f16(pa, vb, o0);
      vb = load_bfrag(VTb + (size_t)(1 * 16 + r) * SEQ + j0 + 16 * g); o1 = wmma_f16(pa, vb, o1);
      vb = load_bfrag(VTb + (size_t)(2 * 16 + r) * SEQ + j0 + 16 * g); o2 = wmma_f16(pa, vb, o2);
      vb = load_bfrag(VTb + (size_t)(3 * 16 + r) * SEQ + j0 + 16 * g); o3 = wmma_f16(pa, vb, o3);
    }
  }

  // Normalize rows and store f16 [TOK, 1024] at column h*64
  half_t* outb = attn_out + ((size_t)b * SEQ + q0) * DIM + h * DHEAD;
#pragma unroll
  for (int v = 0; v < 8; ++v) {
    float inv = 1.0f / rl[v];
    int m = v + 8 * g;
    half_t* orow = outb + (size_t)m * DIM;
    orow[0 * 16 + r] = (half_t)(o0[v] * inv);
    orow[1 * 16 + r] = (half_t)(o1[v] * inv);
    orow[2 * 16 + r] = (half_t)(o2[v] * inv);
    orow[3 * 16 + r] = (half_t)(o3[v] * inv);
  }
}

// ---------------------------------------------------------------------------
// Host launcher
// ---------------------------------------------------------------------------
extern "C" void kernel_launch(void* const* d_in, const int* in_sizes, int n_in,
                              void* d_out, int out_size, void* d_ws, size_t ws_size,
                              hipStream_t stream) {
  (void)in_sizes; (void)n_in; (void)out_size; (void)ws_size;
  const float* x     = (const float*)d_in[0];
  const float* w_qkv = (const float*)d_in[1];
  const float* w_out = (const float*)d_in[2];
  const float* b_out = (const float*)d_in[3];
  const float* g1    = (const float*)d_in[4];
  const float* bt1   = (const float*)d_in[5];
  const float* g2    = (const float*)d_in[6];
  const float* bt2   = (const float*)d_in[7];
  const float* w1    = (const float*)d_in[8];
  const float* b1    = (const float*)d_in[9];
  const float* w2    = (const float*)d_in[10];
  const float* b2    = (const float*)d_in[11];
  float* out = (float*)d_out;

  char* ws = (char*)d_ws;
  size_t off = 0;
  auto salloc = [&](size_t bytes) -> void* {
    void* p = ws + off;
    off = (off + bytes + 255) & ~(size_t)255;
    return p;
  };
  half_t* wqkvT = (half_t*)salloc((size_t)QKV3 * DIM * 2);
  half_t* woutT = (half_t*)salloc((size_t)DIM * DIM * 2);
  half_t* w1T   = (half_t*)salloc((size_t)DIM * DIM * 2);
  half_t* w2T   = (half_t*)salloc((size_t)DIM * DIM * 2);
  half_t* xlh   = (half_t*)salloc((size_t)TOK * DIM * 2);
  half_t* qkvh  = (half_t*)salloc((size_t)TOK * QKV3 * 2);
  half_t* vT    = (half_t*)salloc((size_t)TOK * DIM * 2);
  half_t* attnh = (half_t*)salloc((size_t)TOK * DIM * 2);
  float*  res2  = (float*)salloc((size_t)TOK * DIM * 4);
  half_t* h2h   = (half_t*)salloc((size_t)TOK * DIM * 2);
  half_t* ffh   = (half_t*)salloc((size_t)TOK * DIM * 2);

  // Weights -> f16, transposed to [N,K]
  {
    size_t t = (size_t)DIM * QKV3;
    convT_kernel<<<dim3((unsigned)((t + 255) / 256)), dim3(256), 0, stream>>>(w_qkv, wqkvT, DIM, QKV3);
    t = (size_t)DIM * DIM;
    unsigned gb = (unsigned)((t + 255) / 256);
    convT_kernel<<<dim3(gb), dim3(256), 0, stream>>>(w_out, woutT, DIM, DIM);
    convT_kernel<<<dim3(gb), dim3(256), 0, stream>>>(w1,    w1T,   DIM, DIM);
    convT_kernel<<<dim3(gb), dim3(256), 0, stream>>>(w2,    w2T,   DIM, DIM);
  }

  // LN1 -> xl (f16)
  ln_kernel<<<dim3(TOK), dim3(256), 0, stream>>>(x, g1, bt1, xlh);

  // qkv = xl @ w_qkv  (f16 out)
  gemm_kernel<<<dim3(QKV3 / 128, TOK / 128), dim3(128), 0, stream>>>(
      xlh, wqkvT, nullptr, qkvh, nullptr, nullptr, TOK, QKV3, DIM, 0);

  // V -> [b,h,d,key] for contiguous B-fragments in P*V
  vtrans_kernel<<<dim3((unsigned)(((size_t)TOK * DIM + 255) / 256)), dim3(256), 0, stream>>>(qkvh, vT);

  // attention (4096 wave-tasks, 4 per block)
  attn_kernel<<<dim3(1024), dim3(128), 0, stream>>>(qkvh, vT, attnh);

  // res2 = attn @ w_out + b_out + x  (f32 out)
  gemm_kernel<<<dim3(DIM / 128, TOK / 128), dim3(128), 0, stream>>>(
      attnh, woutT, res2, nullptr, b_out, x, TOK, DIM, DIM, 0);

  // LN2 -> h (f16)
  ln_kernel<<<dim3(TOK), dim3(256), 0, stream>>>(res2, g2, bt2, h2h);

  // ff = gelu(h @ w1 + b1)  (f16 out)
  gemm_kernel<<<dim3(DIM / 128, TOK / 128), dim3(128), 0, stream>>>(
      h2h, w1T, nullptr, ffh, b1, nullptr, TOK, DIM, DIM, 1);

  // out = ff @ w2 + b2 + res2  (f32 out)
  gemm_kernel<<<dim3(DIM / 128, TOK / 128), dim3(128), 0, stream>>>(
      ffh, w2T, out, nullptr, b2, res2, TOK, DIM, DIM, 0);
}
